// DualGNN_31327491457689
// MI455X (gfx1250) — compile-verified, matrix operand order checked by reading, not run on hardware
//
#include <hip/hip_runtime.h>
#include <hip/hip_bf16.h>

// ---------------------------------------------------------------------------
// DualGNN on MI455X (gfx1250).  Memory-bound edge scatter in f32; all dense
// GEMMs on v_wmma_f32_16x16x32_f16 (f16 inputs, f32 accum).  wave32.
// Workspace use: ~72 MB.
// ---------------------------------------------------------------------------

typedef __attribute__((ext_vector_type(16))) _Float16 v16h;
typedef __attribute__((ext_vector_type(8)))  _Float16 v8h;
typedef __attribute__((ext_vector_type(8)))  float    v8f;

#define EPS 1e-5f

// ---------------- WMMA GEMM:  C[M,ldc] = A[M,lda] * Bt^T ------------------
// A  : row-major f16, leading dim lda (multiple of 32 halves -> 16B aligned)
// Bt : weights stored TRANSPOSED, row c holds column c of W, leading dim ldb
// Fragment layout per CDNA5 ISA 7.12.2 (16-bit A 16x32):
//   lanes 0-15 : row = m0+lane,    k = kb+{0..7} and kb+{16..23}
//   lanes 16-31: row = m0+lane-16, k = kb+{8..15} and kb+{24..31}
// B uses the same packing with "row" -> output column (Bt is column-major W).
__device__ __forceinline__ v16h load_frag(const _Float16* __restrict__ base,
                                          int stride, int major0, int limit,
                                          int kb, int lane) {
  int idx = major0 + (lane & 15);
  idx = idx < limit ? idx : limit - 1;            // clamp (stay in-bounds)
  const _Float16* p = base + (size_t)idx * stride + kb + ((lane & 16) ? 8 : 0);
  v8h lo = *(const v8h*)(p);
  v8h hi = *(const v8h*)(p + 16);
  v16h r;
#pragma unroll
  for (int i = 0; i < 8; ++i) { r[i] = lo[i]; r[8 + i] = hi[i]; }
  return r;
}

__global__ void k_wmma_gemm(const _Float16* __restrict__ A, int lda,
                            const _Float16* __restrict__ Bt, int ldb,
                            float* __restrict__ C, int ldc,
                            int M, int ntiles, int ksteps) {
  int wave = (int)((blockIdx.x * blockDim.x + threadIdx.x) >> 5);
  int lane = threadIdx.x & 31;
  int mtiles = (M + 15) >> 4;
  if (wave >= mtiles * ntiles) return;            // wave-uniform: EXEC stays ~0
  int mt = wave / ntiles, nt = wave % ntiles;
  int m0 = mt * 16, n0 = nt * 16;
  v8f acc = {};
  for (int ks = 0; ks < ksteps; ++ks) {
    v16h a = load_frag(A,  lda, m0, M,  ks * 32, lane);
    v16h b = load_frag(Bt, ldb, n0, ntiles * 16, ks * 32, lane);
    acc = __builtin_amdgcn_wmma_f32_16x16x32_f16(
        false, a, false, b, (short)0, acc, false, false);
  }
  // C/D layout: lanes 0-15 -> M=m0+j, N=lane ; lanes 16-31 -> M=m0+8+j
  int col = n0 + (lane & 15);
  int r0  = m0 + ((lane & 16) ? 8 : 0);
#pragma unroll
  for (int j = 0; j < 8; ++j) {
    int r = r0 + j;
    if (r < M) C[(size_t)r * ldc + col] = acc[j];
  }
}

// ---------------- small utility kernels -----------------------------------
__global__ void k_fill(float* p, float v, int n) {
  int i = blockIdx.x * blockDim.x + threadIdx.x;
  if (i < n) p[i] = v;
}

__global__ void k_deg(const int* __restrict__ dst, int E, float* deg) {
  int e = blockIdx.x * blockDim.x + threadIdx.x;
  if (e < E) atomicAdd(&deg[dst[e]], 1.0f);
}

__global__ void k_cnt(const int* __restrict__ batch, int N, float* cnt) {
  int i = blockIdx.x * blockDim.x + threadIdx.x;
  if (i < N) atomicAdd(&cnt[batch[i]], 1.0f);
}

// transpose + zero-pad weight  W[Ksrc,64] -> Wt[64,Kpad] (f16)
__global__ void k_wconv(const float* __restrict__ W, int Ksrc, int Kpad,
                        _Float16* __restrict__ Wt) {
  int i = blockIdx.x * blockDim.x + threadIdx.x;
  if (i >= 64 * Kpad) return;
  int c = i / Kpad, k = i - c * Kpad;
  Wt[i] = (k < Ksrc) ? (_Float16)W[k * 64 + c] : (_Float16)0.f;
}

// pad x[N,7] -> x16[N,32] f16
__global__ void k_padx(const float* __restrict__ x, int N,
                       _Float16* __restrict__ x16) {
  int i = blockIdx.x * blockDim.x + threadIdx.x;
  if (i >= N * 32) return;
  int r = i >> 5, k = i & 31;
  x16[i] = (k < 7) ? (_Float16)x[r * 7 + k] : (_Float16)0.f;
}

// y = xw * dinv(row) in place; zero agg; zero BN accumulators
__global__ void k_scale(float* __restrict__ xw, float* __restrict__ agg,
                        const float* __restrict__ deg, int N,
                        float* __restrict__ bnacc) {
  int i = blockIdx.x * blockDim.x + threadIdx.x;
  if (i < 128) bnacc[i] = 0.f;
  if (i >= N * 64) return;
  int r = i >> 6;
  xw[i] *= rsqrtf(deg[r]);
  agg[i] = 0.f;
}

// scatter-add messages: agg[dst] += y[src]   (16 threads x float4 per edge)
__global__ void k_edges(const int* __restrict__ src, const int* __restrict__ dst,
                        int E, const float* __restrict__ y,
                        float* __restrict__ agg) {
  unsigned gtid = blockIdx.x * blockDim.x + threadIdx.x;
  int e = gtid >> 4;
  if (e >= E) return;
  int c = (gtid & 15) * 4;
  int s = src[e], d = dst[e];
  float4 v = *(const float4*)(y + (size_t)s * 64 + c);
  float* a = agg + (size_t)d * 64 + c;
  atomicAdd(a + 0, v.x); atomicAdd(a + 1, v.y);
  atomicAdd(a + 2, v.z); atomicAdd(a + 3, v.w);
}

// out = dinv*(agg + y) + b  (in place on agg) + BN partial sums (LDS reduce)
__global__ void k_finalize(float* __restrict__ agg, const float* __restrict__ y,
                           const float* __restrict__ deg,
                           const float* __restrict__ bias, int N,
                           float* __restrict__ bnacc) {
  __shared__ float sdata[256];
  int f = threadIdx.x & 63;
  int rstride = (gridDim.x * blockDim.x) >> 6;
  int r = (blockIdx.x * blockDim.x + threadIdx.x) >> 6;
  float s = 0.f, ss = 0.f;
  for (; r < N; r += rstride) {
    size_t idx = (size_t)r * 64 + f;
    float o = rsqrtf(deg[r]) * (agg[idx] + y[idx]) + bias[f];
    agg[idx] = o;
    s += o; ss += o * o;
  }
  sdata[threadIdx.x] = s;
  __syncthreads();
  if (threadIdx.x < 64)
    atomicAdd(&bnacc[f], sdata[threadIdx.x] + sdata[threadIdx.x + 64] +
                         sdata[threadIdx.x + 128] + sdata[threadIdx.x + 192]);
  __syncthreads();
  sdata[threadIdx.x] = ss;
  __syncthreads();
  if (threadIdx.x < 64)
    atomicAdd(&bnacc[64 + f], sdata[threadIdx.x] + sdata[threadIdx.x + 64] +
                              sdata[threadIdx.x + 128] + sdata[threadIdx.x + 192]);
}

__global__ void k_bnfinal(const float* __restrict__ bnacc,
                          const float* __restrict__ g,
                          const float* __restrict__ beta, int N,
                          float* __restrict__ sc, float* __restrict__ sh) {
  int f = threadIdx.x;
  if (f >= 64) return;
  float inv = 1.0f / (float)N;
  float mean = bnacc[f] * inv;
  float var  = bnacc[64 + f] * inv - mean * mean;
  float s = g[f] * rsqrtf(var + EPS);
  sc[f] = s;
  sh[f] = beta[f] - mean * s;
}

// h = relu(out*sc+sh); write f16 copy; optionally pool (sum) per graph
__global__ void k_apply(const float* __restrict__ out,
                        const float* __restrict__ sc, const float* __restrict__ sh,
                        _Float16* __restrict__ h16, int N, int do_pool,
                        const int* __restrict__ batch, float* __restrict__ pool,
                        int cbase) {
  int i = blockIdx.x * blockDim.x + threadIdx.x;
  if (i >= N * 64) return;
  int r = i >> 6, f = i & 63;
  float h = fmaxf(out[i] * sc[f] + sh[f], 0.f);
  h16[i] = (_Float16)h;
  if (do_pool) atomicAdd(&pool[(size_t)batch[r] * 128 + cbase + f], h);
}

// p16 = f16( pool / max(cnt,1) )   (cnt_c for cols 0..63, cnt_s for 64..127)
__global__ void k_pooldiv(const float* __restrict__ pool,
                          const float* __restrict__ cnt,
                          _Float16* __restrict__ p16) {
  int i = blockIdx.x * blockDim.x + threadIdx.x;
  if (i >= 1024 * 128) return;
  int g = i >> 7, j = i & 127;
  float c = (j < 64) ? cnt[g] : cnt[1024 + g];
  p16[i] = (_Float16)(pool[i] / fmaxf(c, 1.0f));
}

__global__ void k_fc1post(float* __restrict__ hh, const float* __restrict__ b) {
  int i = blockIdx.x * blockDim.x + threadIdx.x;
  if (i >= 1024 * 64) return;
  hh[i] = fmaxf(hh[i] + b[i & 63], 0.f);
}

__global__ void k_fc2(const float* __restrict__ hh, const float* __restrict__ W,
                      const float* __restrict__ b, float* __restrict__ out) {
  int i = blockIdx.x * blockDim.x + threadIdx.x;
  if (i >= 2048) return;
  int g = i >> 1, o = i & 1;
  float s = b[o];
#pragma unroll 8
  for (int k = 0; k < 64; ++k) s += hh[g * 64 + k] * W[k * 2 + o];
  out[i] = s;
}

// ---------------------------------------------------------------------------
extern "C" void kernel_launch(void* const* d_in, const int* in_sizes, int n_in,
                              void* d_out, int out_size, void* d_ws, size_t ws_size,
                              hipStream_t stream) {
  const int NC = in_sizes[0] / 7;
  const int EC = in_sizes[1] / 2;
  const int NS = in_sizes[3] / 7;
  const int ES = in_sizes[4] / 2;
  const int G  = 1024;

  const float* x_c  = (const float*)d_in[0];
  const int*   ei_c = (const int*)d_in[1];
  const int*   bat_c= (const int*)d_in[2];
  const float* x_s  = (const float*)d_in[3];
  const int*   ei_s = (const int*)d_in[4];
  const int*   bat_s= (const int*)d_in[5];
  const float* Wc0 = (const float*)d_in[6],  *bc0 = (const float*)d_in[7];
  const float* gc0 = (const float*)d_in[8],  *bec0= (const float*)d_in[9];
  const float* Wc1 = (const float*)d_in[10], *bc1 = (const float*)d_in[11];
  const float* gc1 = (const float*)d_in[12], *bec1= (const float*)d_in[13];
  const float* Ws0 = (const float*)d_in[14], *bs0 = (const float*)d_in[15];
  const float* gs0 = (const float*)d_in[16], *bes0= (const float*)d_in[17];
  const float* Ws1 = (const float*)d_in[18], *bs1 = (const float*)d_in[19];
  const float* gs1 = (const float*)d_in[20], *bes1= (const float*)d_in[21];
  const float* Wf1 = (const float*)d_in[22], *bf1 = (const float*)d_in[23];
  const float* Wf2 = (const float*)d_in[24], *bf2 = (const float*)d_in[25];
  float* outp = (float*)d_out;

  // ------ workspace carve-out (256B aligned slabs) ------
  size_t off = 0;
  auto carve = [&](size_t bytes) -> char* {
    char* p = (char*)d_ws + off;
    off += (bytes + 255) & ~(size_t)255;
    return p;
  };
  float*     deg_c = (float*)carve((size_t)NC * 4);
  float*     deg_s = (float*)carve((size_t)NS * 4);
  float*     bufA  = (float*)carve((size_t)NC * 64 * 4);   // agg -> out
  float*     bufC  = (float*)carve((size_t)NC * 64 * 4);   // xw -> y
  _Float16*  bufB  = (_Float16*)carve((size_t)NC * 64 * 2);// h16 (next GEMM in)
  _Float16*  x16   = (_Float16*)carve((size_t)NC * 32 * 2);
  _Float16*  WtC0  = (_Float16*)carve(64 * 32 * 2);
  _Float16*  WtC1  = (_Float16*)carve(64 * 64 * 2);
  _Float16*  WtS0  = (_Float16*)carve(64 * 32 * 2);
  _Float16*  WtS1  = (_Float16*)carve(64 * 64 * 2);
  _Float16*  WtF1  = (_Float16*)carve(64 * 128 * 2);
  float*     bnacc = (float*)carve(128 * 4);
  float*     bnsc  = (float*)carve(64 * 4);
  float*     bnsh  = (float*)carve(64 * 4);
  float*     pool  = (float*)carve((size_t)G * 128 * 4);
  float*     cnt   = (float*)carve((size_t)2 * G * 4);     // [0..G) chromo, [G..2G) solvent
  _Float16*  p16   = (_Float16*)carve((size_t)G * 128 * 2);
  float*     hh    = (float*)carve((size_t)G * 64 * 4);
  (void)ws_size; (void)n_in; (void)out_size;

  auto cdiv = [](int a, int b) { return (a + b - 1) / b; };

  // ------ setup: degrees (self-loop = 1), graph counts, f16 weights ------
  k_fill<<<cdiv(NC, 256), 256, 0, stream>>>(deg_c, 1.0f, NC);
  k_fill<<<cdiv(NS, 256), 256, 0, stream>>>(deg_s, 1.0f, NS);
  k_fill<<<cdiv(G * 128, 256), 256, 0, stream>>>(pool, 0.0f, G * 128);
  k_fill<<<cdiv(2 * G, 256), 256, 0, stream>>>(cnt, 0.0f, 2 * G);
  k_deg<<<cdiv(EC, 256), 256, 0, stream>>>(ei_c + EC, EC, deg_c);
  k_deg<<<cdiv(ES, 256), 256, 0, stream>>>(ei_s + ES, ES, deg_s);
  k_cnt<<<cdiv(NC, 256), 256, 0, stream>>>(bat_c, NC, cnt);
  k_cnt<<<cdiv(NS, 256), 256, 0, stream>>>(bat_s, NS, cnt + G);
  k_wconv<<<cdiv(64 * 32, 256), 256, 0, stream>>>(Wc0, 7, 32, WtC0);
  k_wconv<<<cdiv(64 * 64, 256), 256, 0, stream>>>(Wc1, 64, 64, WtC1);
  k_wconv<<<cdiv(64 * 32, 256), 256, 0, stream>>>(Ws0, 7, 32, WtS0);
  k_wconv<<<cdiv(64 * 64, 256), 256, 0, stream>>>(Ws1, 64, 64, WtS1);
  k_wconv<<<cdiv(64 * 128, 256), 256, 0, stream>>>(Wf1, 128, 128, WtF1);

  // ------ one GCN layer + BN/ReLU ------
  auto gcn_layer = [&](const _Float16* Ain, int lda, const _Float16* Wt, int ldb,
                       int ksteps, int N, int E, const int* ei, const float* deg,
                       const float* bias, const float* gam, const float* bet,
                       int do_pool, const int* batch, int cbase) {
    int waves = ((N + 15) / 16) * 4;
    k_wmma_gemm<<<cdiv(waves, 8), 256, 0, stream>>>(Ain, lda, Wt, ldb,
                                                    bufC, 64, N, 4, ksteps);
    k_scale<<<cdiv(N * 64, 256), 256, 0, stream>>>(bufC, bufA, deg, N, bnacc);
    k_edges<<<cdiv(E * 16, 256), 256, 0, stream>>>(ei, ei + E, E, bufC, bufA);
    k_finalize<<<512, 256, 0, stream>>>(bufA, bufC, deg, bias, N, bnacc);
    k_bnfinal<<<1, 64, 0, stream>>>(bnacc, gam, bet, N, bnsc, bnsh);
    k_apply<<<cdiv(N * 64, 256), 256, 0, stream>>>(bufA, bnsc, bnsh, bufB, N,
                                                   do_pool, batch, pool, cbase);
  };

  // ------ chromo branch ------
  k_padx<<<cdiv(NC * 32, 256), 256, 0, stream>>>(x_c, NC, x16);
  gcn_layer(x16, 32, WtC0, 32, 1, NC, EC, ei_c, deg_c, bc0, gc0, bec0, 0, bat_c, 0);
  gcn_layer(bufB, 64, WtC1, 64, 2, NC, EC, ei_c, deg_c, bc1, gc1, bec1, 1, bat_c, 0);

  // ------ solvent branch ------
  k_padx<<<cdiv(NS * 32, 256), 256, 0, stream>>>(x_s, NS, x16);
  gcn_layer(x16, 32, WtS0, 32, 1, NS, ES, ei_s, deg_s, bs0, gs0, bes0, 0, bat_s, 0);
  gcn_layer(bufB, 64, WtS1, 64, 2, NS, ES, ei_s, deg_s, bs1, gs1, bes1, 1, bat_s, 64);

  // ------ head: mean pool -> fc1 (WMMA) -> relu -> fc2 ------
  k_pooldiv<<<cdiv(G * 128, 256), 256, 0, stream>>>(pool, cnt, p16);
  {
    int waves = (G / 16) * 4;
    k_wmma_gemm<<<cdiv(waves, 8), 256, 0, stream>>>(p16, 128, WtF1, 128,
                                                    hh, 64, G, 4, 4);
  }
  k_fc1post<<<cdiv(G * 64, 256), 256, 0, stream>>>(hh, bf1);
  k_fc2<<<cdiv(2 * G, 256), 256, 0, stream>>>(hh, Wf2, bf2, outp);
}
